// RetNet_31636729102545
// MI455X (gfx1250) — matrix operand order repaired and checked by reading, not compiled, not run on hardware
//
#include <hip/hip_runtime.h>
#include <hip/hip_bf16.h>
#include <math.h>

// ---------------- problem constants ----------------
#define LAYERS 2
#define DIM    1024
#define FFN_D  4096
#define HEADS  8
#define HD     128
#define HV     128
#define VD     1024
#define BB     2
#define LL     2048
#define BL     (BB*LL)
#define EPSF   1e-5f

#define GAS __attribute__((address_space(1)))
#define LAS __attribute__((address_space(3)))

// ---------------- CDNA5 feature detection (compile-safe fallbacks) ----------
#if defined(__has_builtin)
# if __has_builtin(__builtin_amdgcn_global_load_async_to_lds_b128)
#  define HAVE_ASYNC 1
# endif
# if __has_builtin(__builtin_amdgcn_tensor_load_to_lds)
#  define HAVE_TDM 1
# endif
#endif

#if defined(__has_builtin) && __has_builtin(__builtin_amdgcn_s_wait_asynccnt)
# define WAIT_ASYNC() __builtin_amdgcn_s_wait_asynccnt(0)
#else
# define WAIT_ASYNC() asm volatile("s_wait_asynccnt 0" ::: "memory")
#endif

typedef unsigned int u32x4 __attribute__((ext_vector_type(4)));
typedef int          i32x4 __attribute__((ext_vector_type(4)));
typedef int          i32x8 __attribute__((ext_vector_type(8)));
typedef int          v4i   __attribute__((ext_vector_type(4)));

#if defined(HAVE_ASYNC)
// signature (from clang diagnostic): (AS1 v4i*, AS3 v4i*, imm offset, imm cpol)
# define ASYNC_LD_B128(gptr, lptr) \
    __builtin_amdgcn_global_load_async_to_lds_b128( \
        (GAS v4i*)(size_t)(gptr), (LAS v4i*)(LAS void*)(lptr), 0, 0)
#endif

#if defined(HAVE_TDM)
# if __has_include(<hip/amd_detail/amd_gfx1250_TDM.h>)
   // amdgpu-toolchain / therock headers -> 6-arg builtin
#  define TDM_LOAD(g0,g1,g2,g3) do { i32x8 _z8 = (i32x8)0; \
      __builtin_amdgcn_tensor_load_to_lds((g0),(g1),(g2),(g3),_z8,0); } while(0)
# else
   // ROCm 7.2 -> 5-arg builtin
#  define TDM_LOAD(g0,g1,g2,g3) __builtin_amdgcn_tensor_load_to_lds((g0),(g1),(g2),(g3),0)
# endif
#endif

// ---------------- WMMA types / helpers ----------------
typedef __bf16 v16bf __attribute__((ext_vector_type(16)));
typedef float  v8f   __attribute__((ext_vector_type(8)));

union AB16 { v16bf v; unsigned int u[8]; unsigned short s[16]; };
union CD8  { v8f v; float f[8]; };

static __device__ __forceinline__ unsigned short f2bf(float x) {
  unsigned int u = __float_as_uint(x);
  unsigned int r = ((u >> 16) & 1u) + 0x7FFFu;   // round-to-nearest-even
  return (unsigned short)((u + r) >> 16);
}

static __device__ __forceinline__ v8f wmma_bf16(v16bf a, v16bf b, v8f c) {
  return __builtin_amdgcn_wmma_f32_16x16x32_bf16(false, a, false, b, (short)0, c, false, false);
}

// A-matrix (16x32, 16-bit) per-lane K index for dword d (contiguous: k = 16*half + 2*d, matches B)
static __device__ __forceinline__ int a_kidx(int d, int half) {
  return 16*half + 2*d;
}

// ---------------- generic bf16 GEMM, 128x128 tile, 8 waves ----------------
__global__ __launch_bounds__(256) void gemm_bf16_kernel(
    const unsigned short* __restrict__ A, const unsigned short* __restrict__ Bm,
    float* __restrict__ Cf, unsigned short* __restrict__ Cb,
    const float* __restrict__ bias, const float* __restrict__ resid,
    int M, int N, int K, int act)
{
  __shared__ unsigned short Al[128*32];   // [m][k]
  __shared__ unsigned short Blds[128*32]; // transposed: [n][k]
  const int tid  = threadIdx.x;
  const int lane = tid & 31;
  const int w    = tid >> 5;
  const int m0   = blockIdx.y * 128;
  const int n0   = blockIdx.x * 128;
  const int half = lane >> 4;
  const int lm   = lane & 15;

  CD8 acc[8];
  #pragma unroll
  for (int j = 0; j < 8; j++)
    #pragma unroll
    for (int q = 0; q < 8; q++) acc[j].f[q] = 0.0f;

  const int ar = tid >> 1;         // 0..127 (A row)
  const int ac = (tid & 1) * 16;   // 0/16
  const int bk = tid >> 3;         // 0..31  (B row = k)
  const int bn = (tid & 7) * 16;   // 0..112

  for (int k0 = 0; k0 < K; k0 += 32) {
    // A tile 128x32 -> LDS (async DMA path when available)
    const unsigned short* ag = A + (size_t)(m0 + ar) * K + k0 + ac;
#if defined(HAVE_ASYNC)
    ASYNC_LD_B128(ag,     &Al[ar*32 + ac]);
    ASYNC_LD_B128(ag + 8, &Al[ar*32 + ac + 8]);
#else
    *(uint4*)&Al[ar*32 + ac]     = *(const uint4*)(ag);
    *(uint4*)&Al[ar*32 + ac + 8] = *(const uint4*)(ag + 8);
#endif
    // B tile 32x128 -> transposed [n][k] (needs lane transpose; stays manual)
    const unsigned short* bg = Bm + (size_t)(k0 + bk) * N + n0 + bn;
    unsigned short tb[16];
    *(uint4*)&tb[0] = *(const uint4*)(bg);
    *(uint4*)&tb[8] = *(const uint4*)(bg + 8);
    #pragma unroll
    for (int j = 0; j < 16; j++) Blds[(bn + j)*32 + bk] = tb[j];
    if (k0 + 32 < K) {                       // -> global_prefetch_b8
      __builtin_prefetch(ag + 32, 0, 0);
      __builtin_prefetch(bg + (size_t)32 * N, 0, 0);
    }
#if defined(HAVE_ASYNC)
    WAIT_ASYNC();
#endif
    __syncthreads();

    AB16 af;
    #pragma unroll
    for (int d = 0; d < 8; d++)
      af.u[d] = *(const unsigned int*)&Al[(w*16 + lm)*32 + a_kidx(d, half)];
    #pragma unroll
    for (int j = 0; j < 8; j++) {
      AB16 bf_;
      #pragma unroll
      for (int d = 0; d < 8; d++)
        bf_.u[d] = *(const unsigned int*)&Blds[(j*16 + lm)*32 + 16*half + 2*d];
      acc[j].v = wmma_bf16(af.v, bf_.v, acc[j].v);
    }
    __syncthreads();
  }

  #pragma unroll
  for (int j = 0; j < 8; j++)
    #pragma unroll
    for (int v = 0; v < 8; v++) {
      int row = m0 + w*16 + v + 8*half;
      int col = n0 + j*16 + lm;
      float x = acc[j].f[v];
      if (bias)  x += bias[col];
      if (act)   x = 0.5f * x * (1.0f + erff(x * 0.70710678118654752f)); // exact GELU
      if (resid) x += resid[(size_t)row * N + col];
      if (Cf) Cf[(size_t)row * N + col] = x;
      if (Cb) Cb[(size_t)row * N + col] = f2bf(x);
    }
}

// ---------------- fused causal retention: Y = (Q K^T (.) D) V ----------------
// grid: (LL/64, B*HEADS), block 128 (4 waves; wave owns 16 rows)
__global__ __launch_bounds__(128) void retnet_attn_kernel(
    const unsigned short* __restrict__ Qb, const unsigned short* __restrict__ Kb,
    const unsigned short* __restrict__ Vb, float* __restrict__ Y)
{
  __shared__ unsigned short Kl[32*128];     // [m][hd]  (== [n][k] for B frags)
  __shared__ unsigned short Vl[128*32];     // transposed [hv][m]
  __shared__ unsigned short Sl[4][16*32];   // per-wave decayed score tile (bf16)
  const int tid = threadIdx.x, lane = tid & 31, w = tid >> 5;
  const int bh = blockIdx.y;
  const int h  = bh & (HEADS-1);
  const int r0 = blockIdx.x*64 + w*16;
  const int half = lane >> 4, lm = lane & 15;

  // gammas = 1 - exp(linspace(log(1/32), log(1/512), HEADS))
  const float lg0 = logf(1.0f/32.0f), lg1 = logf(1.0f/512.0f);
  const float gamma = 1.0f - expf(lg0 + (lg1 - lg0) * ((float)h / (float)(HEADS-1)));
  const float logg = logf(gamma);

  // preload Q fragments (16x128 = 4 chunks of 16x32)
  const unsigned short* Qp = Qb + ((size_t)bh * LL + r0 + lm) * HD;
  AB16 qf[4];
  #pragma unroll
  for (int c = 0; c < 4; c++)
    #pragma unroll
    for (int d = 0; d < 8; d++)
      qf[c].u[d] = *(const unsigned int*)(Qp + c*32 + a_kidx(d, half));

  CD8 acc[8];
  #pragma unroll
  for (int j = 0; j < 8; j++)
    #pragma unroll
    for (int q = 0; q < 8; q++) acc[j].f[q] = 0.0f;

  const int nch = blockIdx.x*2 + 2;   // causal: chunks of 32 covering m <= max row
  for (int mc = 0; mc < nch; mc++) {
    const int m0 = mc * 32;
    const int kr = tid >> 2, cb = (tid & 3) * 32;

    // ---- K tile 32x128 -> LDS ----
#if defined(HAVE_TDM)
    if (w == 0) {
      // Tensor DMA: 2D tile (tile_dim0=128 elems/row, tile_dim1=32 rows),
      // tensor 128 x 2048 bf16, row stride 128; dest = Kl (contiguous).
      unsigned long long ga = (unsigned long long)(size_t)(Kb + ((size_t)bh*LL + m0)*HD);
      unsigned ldsoff = (unsigned)(size_t)(LAS unsigned short*)Kl;
      u32x4 g0;
      g0[0] = 1u;                                  // count=1 (valid descriptor)
      g0[1] = ldsoff;                              // lds_addr
      g0[2] = (unsigned)ga;                        // global_addr[31:0]
      g0[3] = ((unsigned)(ga >> 32) & 0x01FFFFFFu) | 0x80000000u; // addr[56:32], type=2
      i32x8 g1;
      g1[0] = 1 << 16;                             // data_size = 2 bytes
      g1[1] = (int)(HD << 16);                     // tensor_dim0 = 128
      g1[2] = (int)(LL << 16);                     // tensor_dim1 = 2048
      g1[3] = (int)(HD << 16);                     // tile_dim0 = 128
      g1[4] = 32;                                  // tile_dim1 = 32 (tile_dim2 = 0)
      g1[5] = HD;                                  // tensor_dim0_stride = 128
      g1[6] = 0; g1[7] = 0;                        // dim1 stride unused (2D)
      i32x4 gz = (i32x4)0;
      TDM_LOAD(g0, g1, gz, gz);
      __builtin_amdgcn_s_wait_tensorcnt(0);
    }
#elif defined(HAVE_ASYNC)
    {
      const unsigned short* kg = Kb + ((size_t)bh*LL + m0 + kr)*HD + cb;
      #pragma unroll
      for (int q = 0; q < 4; q++)
        ASYNC_LD_B128(kg + q*8, &Kl[kr*128 + cb + q*8]);
    }
#else
    {
      const unsigned short* kg = Kb + ((size_t)bh*LL + m0 + kr)*HD + cb;
      #pragma unroll
      for (int q = 0; q < 4; q++)
        *(uint4*)&Kl[kr*128 + cb + q*8] = *(const uint4*)(kg + q*8);
    }
#endif
    // ---- V tile 32x128 -> transposed LDS (lane transpose; manual) ----
    {
      const unsigned short* vg = Vb + ((size_t)bh*LL + m0 + kr)*HV + cb;
      unsigned short tv[32];
      #pragma unroll
      for (int q = 0; q < 4; q++) *(uint4*)&tv[q*8] = *(const uint4*)(vg + q*8);
      #pragma unroll
      for (int j = 0; j < 32; j++) Vl[(cb + j)*32 + kr] = tv[j];
    }
#if !defined(HAVE_TDM) && defined(HAVE_ASYNC)
    WAIT_ASYNC();
#endif
    __syncthreads();

    // S(16x32) = Q(16x128) * K_tile^T, two 16x16 C tiles
    CD8 s0, s1;
    #pragma unroll
    for (int q = 0; q < 8; q++) { s0.f[q] = 0.0f; s1.f[q] = 0.0f; }
    #pragma unroll
    for (int c = 0; c < 4; c++) {
      AB16 b0, b1;
      #pragma unroll
      for (int d = 0; d < 8; d++) {
        int kk = c*32 + 16*half + 2*d;
        b0.u[d] = *(const unsigned int*)&Kl[lm*128 + kk];
        b1.u[d] = *(const unsigned int*)&Kl[(16 + lm)*128 + kk];
      }
      s0.v = wmma_bf16(qf[c].v, b0.v, s0.v);
      s1.v = wmma_bf16(qf[c].v, b1.v, s1.v);
    }

    // causal decay D(l,m) = exp(logg*(l-m)) for l>=m, stage bf16 A tile
    #pragma unroll
    for (int v = 0; v < 8; v++) {
      int i = v + 8*half;
      float dl0 = (float)(r0 + i) - (float)(m0 + lm);
      float dl1 = dl0 - 16.0f;
      float e0 = (dl0 >= 0.0f) ? expf(logg * dl0) : 0.0f;
      float e1 = (dl1 >= 0.0f) ? expf(logg * dl1) : 0.0f;
      Sl[w][i*32 + lm]      = f2bf(s0.f[v] * e0);
      Sl[w][i*32 + 16 + lm] = f2bf(s1.f[v] * e1);
    }
    asm volatile("s_wait_dscnt 0" ::: "memory");  // intra-wave DS RAW before re-read

    AB16 sa;
    #pragma unroll
    for (int d = 0; d < 8; d++)
      sa.u[d] = *(const unsigned int*)&Sl[w][lm*32 + a_kidx(d, half)];
    #pragma unroll
    for (int j = 0; j < 8; j++) {
      AB16 bv;
      #pragma unroll
      for (int d = 0; d < 8; d++)
        bv.u[d] = *(const unsigned int*)&Vl[(j*16 + lm)*32 + 16*half + 2*d];
      acc[j].v = wmma_bf16(sa.v, bv.v, acc[j].v);
    }
    __syncthreads();
  }

  #pragma unroll
  for (int j = 0; j < 8; j++)
    #pragma unroll
    for (int v = 0; v < 8; v++) {
      int i = v + 8*half;
      Y[((size_t)bh*LL + r0 + i)*HV + j*16 + lm] = acc[j].f[v];
    }
}

// ---------------- LayerNorm (row = b*L), writes bf16 ----------------
__global__ __launch_bounds__(256) void layernorm_kernel(
    const float* __restrict__ X, const float* __restrict__ wgt,
    const float* __restrict__ bia, unsigned short* __restrict__ out)
{
  __shared__ float red[256], red2[256];
  int row = blockIdx.x, tid = threadIdx.x;
  const float* xr = X + (size_t)row * DIM;
  float xv[4], s = 0.f, s2 = 0.f;
  #pragma unroll
  for (int q = 0; q < 4; q++) { float x = xr[tid + q*256]; xv[q] = x; s += x; s2 += x*x; }
  red[tid] = s; red2[tid] = s2; __syncthreads();
  for (int off = 128; off > 0; off >>= 1) {
    if (tid < off) { red[tid] += red[tid+off]; red2[tid] += red2[tid+off]; }
    __syncthreads();
  }
  float mu = red[0] * (1.0f/DIM);
  float var = red2[0] * (1.0f/DIM) - mu*mu;
  float rs = rsqrtf(var + EPSF);
  #pragma unroll
  for (int q = 0; q < 4; q++) {
    int e = tid + q*256;
    out[(size_t)row*DIM + e] = f2bf((xv[q] - mu) * rs * wgt[e] + bia[e]);
  }
}

// ---------------- xPos rotary + head-major pack (f32 -> bf16) ----------------
__global__ void xpos_pack_kernel(const float* __restrict__ src,
                                 unsigned short* __restrict__ dst, int down)
{
  int p = blockIdx.x * blockDim.x + threadIdx.x;      // BB*LL*HEADS*(HD/2)
  if (p >= BB*LL*HEADS*(HD/2)) return;
  int e2 = p & (HD/2 - 1);
  int t  = p / (HD/2);
  int hh = t & (HEADS-1);
  int bl = t / HEADS;
  int l  = bl & (LL-1);
  int b  = bl / LL;
  float x0 = src[(size_t)bl*DIM + hh*HD + 2*e2];
  float x1 = src[(size_t)bl*DIM + hh*HD + 2*e2 + 1];
  float fl = (float)l;
  float base = (2.0f*(float)e2 + 0.4f*(float)HD) / (1.4f*(float)HD);
  float sc = powf(base, fl * (1.0f/512.0f));
  if (down) sc = 1.0f/sc;
  float invf = powf(10000.0f, -(float)e2 / (float)(HD/2));
  float ang = fl * invf;
  float sn = sinf(ang)*sc, cs = cosf(ang)*sc;
  size_t o = (((size_t)(b*HEADS + hh))*LL + l)*HD + 2*e2;
  dst[o]     = f2bf(x0*cs - x1*sn);
  dst[o + 1] = f2bf(x1*cs + x0*sn);
}

// ---------------- V pack (reshape to head-major, bf16) ----------------
__global__ void vpack_kernel(const float* __restrict__ src, unsigned short* __restrict__ dst)
{
  int p = blockIdx.x * blockDim.x + threadIdx.x;      // BB*LL*VD
  if (p >= BB*LL*VD) return;
  int e  = p & (HV-1);
  int t  = p / HV;
  int hh = t & (HEADS-1);
  int bl = t / HEADS;
  int l  = bl & (LL-1);
  int b  = bl / LL;
  dst[(((size_t)(b*HEADS + hh))*LL + l)*HV + e] = f2bf(src[(size_t)bl*DIM + hh*HV + e]);
}

// ---------------- per-head GroupNorm + SiLU gate, writes bf16 ----------------
__global__ __launch_bounds__(128) void gngate_kernel(
    const float* __restrict__ Y, const float* __restrict__ G,
    const float* __restrict__ gw, const float* __restrict__ gb,
    unsigned short* __restrict__ out)
{
  __shared__ float red[128], red2[128];
  int l = blockIdx.x, bh = blockIdx.y;
  int b = bh >> 3, hh = bh & 7;
  int v = threadIdx.x;
  float y = Y[(((size_t)bh)*LL + l)*HV + v];
  red[v] = y; red2[v] = y*y; __syncthreads();
  for (int off = 64; off > 0; off >>= 1) {
    if (v < off) { red[v] += red[v+off]; red2[v] += red2[v+off]; }
    __syncthreads();
  }
  float mu  = red[0] * (1.0f/HV);
  float var = red2[0] * (1.0f/HV) - mu*mu;
  int col = hh*HV + v;
  float yn = (y - mu) * rsqrtf(var + EPSF) * gw[col] + gb[col];
  size_t bl = (size_t)b*LL + l;
  float g = G[bl*VD + col];
  float gate = g / (1.0f + expf(-g));     // g * sigmoid(g)
  out[bl*VD + col] = f2bf(gate * yn);
}

// ---------------- weight conversion ----------------
__global__ void conv_qkv_kernel(const float* __restrict__ Win, unsigned short* __restrict__ Wout)
{
  int p = blockIdx.x * blockDim.x + threadIdx.x;
  if (p >= DIM*DIM) return;
  int col = p & (DIM-1), d = p >> 10;
  int hh = col >> 7, e = col & 127;
  Wout[(size_t)d*DIM + col] = f2bf(Win[((size_t)hh*DIM + d)*HD + e]);
}
__global__ void conv_copy_kernel(const float* __restrict__ Win, unsigned short* __restrict__ Wout, int n)
{
  int p = blockIdx.x * blockDim.x + threadIdx.x;
  if (p < n) Wout[p] = f2bf(Win[p]);
}

// ---------------- host driver ----------------
extern "C" void kernel_launch(void* const* d_in, const int* in_sizes, int n_in,
                              void* d_out, int out_size, void* d_ws, size_t ws_size,
                              hipStream_t stream)
{
  const float* X     = (const float*)d_in[0];
  const float* Wq    = (const float*)d_in[1];
  const float* Wk    = (const float*)d_in[2];
  const float* Wv    = (const float*)d_in[3];
  const float* W_G   = (const float*)d_in[4];
  const float* W_O   = (const float*)d_in[5];
  const float* gn_w  = (const float*)d_in[6];
  const float* gn_b  = (const float*)d_in[7];
  const float* ln1_w = (const float*)d_in[8];
  const float* ln1_b = (const float*)d_in[9];
  const float* ln2_w = (const float*)d_in[10];
  const float* ln2_b = (const float*)d_in[11];
  const float* fw1   = (const float*)d_in[12];
  const float* fb1   = (const float*)d_in[13];
  const float* fw2   = (const float*)d_in[14];
  const float* fb2   = (const float*)d_in[15];

  char* ws = (char*)d_ws;
  size_t off = 0;
  auto alloc = [&](size_t bytes) -> void* {
    off = (off + 255) & ~(size_t)255;
    void* p = ws + off; off += bytes; return p;
  };

  unsigned short* XnB  = (unsigned short*)alloc((size_t)BL*DIM*2);
  float*          LinF = (float*)alloc((size_t)BL*DIM*4);
  float*          Gf   = (float*)alloc((size_t)BL*VD*4);
  unsigned short* Qbf  = (unsigned short*)alloc((size_t)BB*HEADS*LL*HD*2);
  unsigned short* Kbf  = (unsigned short*)alloc((size_t)BB*HEADS*LL*HD*2);
  unsigned short* Vbf  = (unsigned short*)alloc((size_t)BB*HEADS*LL*HV*2);
  float*          Yat  = (float*)alloc((size_t)BB*HEADS*LL*HV*4);
  unsigned short* YgB  = (unsigned short*)alloc((size_t)BL*VD*2);
  float*          Xbuf = (float*)alloc((size_t)BL*DIM*4);
  float*          Ybuf = (float*)alloc((size_t)BL*DIM*4);
  unsigned short* HnB  = (unsigned short*)alloc((size_t)BL*DIM*2);
  unsigned short* H1B  = (unsigned short*)alloc((size_t)BL*FFN_D*2);
  unsigned short* WqB  = (unsigned short*)alloc((size_t)DIM*DIM*2);
  unsigned short* WkB  = (unsigned short*)alloc((size_t)DIM*DIM*2);
  unsigned short* WvB  = (unsigned short*)alloc((size_t)DIM*DIM*2);
  unsigned short* WGB  = (unsigned short*)alloc((size_t)DIM*VD*2);
  unsigned short* WOB  = (unsigned short*)alloc((size_t)VD*DIM*2);
  unsigned short* W1B  = (unsigned short*)alloc((size_t)DIM*FFN_D*2);
  unsigned short* W2B  = (unsigned short*)alloc((size_t)FFN_D*DIM*2);

  auto gemm = [&](const unsigned short* A, const unsigned short* Bm,
                  float* Cf, unsigned short* Cb, const float* bias,
                  const float* resid, int M, int N, int K, int act) {
    dim3 g(N/128, M/128);
    gemm_bf16_kernel<<<g, 256, 0, stream>>>(A, Bm, Cf, Cb, bias, resid, M, N, K, act);
  };

  for (int i = 0; i < LAYERS; i++) {
    const float* Xin = (i == 0) ? X : Xbuf;

    conv_qkv_kernel<<<(DIM*DIM)/256, 256, 0, stream>>>(Wq + (size_t)i*HEADS*DIM*HD, WqB);
    conv_qkv_kernel<<<(DIM*DIM)/256, 256, 0, stream>>>(Wk + (size_t)i*HEADS*DIM*HD, WkB);
    conv_qkv_kernel<<<(DIM*DIM)/256, 256, 0, stream>>>(Wv + (size_t)i*HEADS*DIM*HV, WvB);
    conv_copy_kernel<<<(DIM*VD)/256, 256, 0, stream>>>(W_G + (size_t)i*DIM*VD, WGB, DIM*VD);
    conv_copy_kernel<<<(VD*DIM)/256, 256, 0, stream>>>(W_O + (size_t)i*VD*DIM, WOB, VD*DIM);
    conv_copy_kernel<<<(DIM*FFN_D)/256, 256, 0, stream>>>(fw1 + (size_t)i*DIM*FFN_D, W1B, DIM*FFN_D);
    conv_copy_kernel<<<(FFN_D*DIM)/256, 256, 0, stream>>>(fw2 + (size_t)i*FFN_D*DIM, W2B, FFN_D*DIM);

    layernorm_kernel<<<BL, 256, 0, stream>>>(Xin, ln1_w + (size_t)i*DIM, ln1_b + (size_t)i*DIM, XnB);

    const int xp_blocks = (BB*LL*HEADS*(HD/2)) / 256;
    gemm(XnB, WqB, LinF, nullptr, nullptr, nullptr, BL, DIM, DIM, 0);
    xpos_pack_kernel<<<xp_blocks, 256, 0, stream>>>(LinF, Qbf, 0);
    gemm(XnB, WkB, LinF, nullptr, nullptr, nullptr, BL, DIM, DIM, 0);
    xpos_pack_kernel<<<xp_blocks, 256, 0, stream>>>(LinF, Kbf, 1);
    gemm(XnB, WvB, LinF, nullptr, nullptr, nullptr, BL, DIM, DIM, 0);
    vpack_kernel<<<(BB*LL*VD)/256, 256, 0, stream>>>(LinF, Vbf);
    gemm(XnB, WGB, Gf, nullptr, nullptr, nullptr, BL, VD, DIM, 0);

    {
      dim3 g(LL/64, BB*HEADS);
      retnet_attn_kernel<<<g, 128, 0, stream>>>(Qbf, Kbf, Vbf, Yat);
    }
    {
      dim3 g(LL, BB*HEADS);
      gngate_kernel<<<g, 128, 0, stream>>>(Yat, Gf, gn_w + (size_t)i*VD, gn_b + (size_t)i*VD, YgB);
    }

    gemm(YgB, WOB, Ybuf, nullptr, nullptr, Xin, BL, DIM, VD, 0);

    layernorm_kernel<<<BL, 256, 0, stream>>>(Ybuf, ln2_w + (size_t)i*DIM, ln2_b + (size_t)i*DIM, HnB);
    gemm(HnB, W1B, nullptr, H1B, fb1 + (size_t)i*FFN_D, nullptr, BL, FFN_D, DIM, 1);
    float* Xout = (i == LAYERS-1) ? (float*)d_out : Xbuf;
    gemm(H1B, W2B, Xout, nullptr, fb2 + (size_t)i*DIM, Ybuf, BL, DIM, FFN_D, 0);
  }
}
